// SelectiveStateSpace_50259707297888
// MI455X (gfx1250) — matrix operand order repaired and checked by reading, not compiled, not run on hardware
//
#include <hip/hip_runtime.h>
#include <hip/hip_bf16.h>

// ---- types (plain ext-vectors: safe inside unions, unlike HIP vector classes)
typedef __attribute__((ext_vector_type(16))) __bf16       v16bf;
typedef __attribute__((ext_vector_type(8)))  float        v8f;
typedef __attribute__((ext_vector_type(4)))  unsigned int u32x4;
typedef __attribute__((ext_vector_type(2)))  unsigned int u32x2;
typedef __attribute__((ext_vector_type(4)))  float        f32x4;

#define D_MODEL 1024
#define NSTATE  64
#define BATCH   8
#define SEQ     2048
#define CHUNK   64
#define NCHUNK  (SEQ / CHUNK)          // 32
#define M_TOTAL (BATCH * SEQ)          // 16384
#define M_TILE  32                     // rows of x per workgroup in the GEMM
#define XROW    1032                   // padded LDS row stride (bf16 elems)

// workspace layout (float offsets); each [16384,64] array = 1M floats = 4MB
#define WS_DELTA 0
#define WS_U     (1u * 1048576u)
#define WS_CT    (2u * 1048576u)
#define WS_P     (3u * 1048576u)
#define WS_HLOC  (4u * 1048576u)
#define WS_HST   (5u * 1048576u)
#define WS_WBF   (6u * 1048576u)       // bf16 region: 3 * 64 * 1024 bf16

// -------------------------------------------------------------------------
// CDNA5 async memory->LDS copy (ASYNCcnt-tracked, no VGPR round trip).
// GV mode: per-lane LDS byte offset in one VGPR, per-lane 64-bit global
// address in a VGPR pair.  (CDNA5 ISA)
// -------------------------------------------------------------------------
__device__ __forceinline__ void async_ld_b128(unsigned lds_byte_off,
                                              unsigned long long gaddr) {
  asm volatile("global_load_async_to_lds_b128 %0, %1, off"
               :: "v"(lds_byte_off), "v"(gaddr) : "memory");
}
__device__ __forceinline__ void async_wait_all() {
  asm volatile("s_wait_asynccnt 0" ::: "memory");
}
// flat shared-aperture address -> raw LDS offset (LDS offset is the low 32 bits)
__device__ __forceinline__ unsigned lds_off(const void* p) {
  return (unsigned)(unsigned long long)p;
}

// -------------------------------------------------------------------------
// Kernel 1: convert the three live projection weights f32 -> bf16
// -------------------------------------------------------------------------
__global__ __launch_bounds__(256) void k_cvt_w(const float* __restrict__ wd,
                                               const float* __restrict__ wb,
                                               const float* __restrict__ wc,
                                               __bf16* __restrict__ out) {
  int idx = blockIdx.x * 256 + threadIdx.x;          // 0 .. 196607
  const float* src = (idx < 65536) ? wd : ((idx < 131072) ? wb : wc);
  out[idx] = (__bf16)src[idx & 65535];
}

// -------------------------------------------------------------------------
// Kernel 2: fused projection GEMMs + pointwise epilogues via WMMA bf16
//   grid = 512 (32-row M tiles), block = 384 (12 waves):
//   wave w -> matrix (w/4) in {delta,Bp,Cp}, N-tile (w%4)*16.
//   Each wave keeps TWO accumulators (rows 0-15 and 16-31 of the tile) and
//   reuses each B fragment for two WMMAs -> halves L2 weight traffic.
// -------------------------------------------------------------------------
__global__ __launch_bounds__(384) void k_gemm(
    const float* __restrict__ x, const __bf16* __restrict__ wbf,
    const float* __restrict__ b_delta, const float* __restrict__ b_Bp,
    const float* __restrict__ b_Cp,
    float* __restrict__ ws_delta, float* __restrict__ ws_u,
    float* __restrict__ ws_ct) {
  // x tile in bf16, row stride 1032 elems (2064B): bank-conflict-avoiding pad
  extern __shared__ __align__(16) __bf16 lds_x[];    // M_TILE * XROW

  const int tid = threadIdx.x;
  const int m0  = blockIdx.x * M_TILE;

  // cooperative load + convert: M_TILE rows x 1024 f32 -> bf16 LDS
  for (int idx = tid; idx < M_TILE * 256; idx += 384) {
    const int row = idx >> 8, c4 = idx & 255;
    const f32x4 v = *(const f32x4*)(x + (size_t)(m0 + row) * D_MODEL + c4 * 4);
    union { __bf16 h[4]; u32x2 u2; } pk;
    pk.h[0] = (__bf16)v.x; pk.h[1] = (__bf16)v.y;
    pk.h[2] = (__bf16)v.z; pk.h[3] = (__bf16)v.w;
    *(u32x2*)(&lds_x[row * XROW + c4 * 4]) = pk.u2;
  }
  __syncthreads();

  const int wave  = tid >> 5;
  const int lane  = tid & 31;
  const int mat   = wave >> 2;                 // 0:delta 1:Bp 2:Cp
  const int n0    = (wave & 3) * 16;
  const int lhalf = (lane < 16) ? 0 : 1;
  const int l15   = lane & 15;

  // B fragment source: column n of B == row (n0+n) of W, contiguous K
  const __bf16* wrow = wbf + mat * 65536 + (size_t)(n0 + l15) * D_MODEL;

  v8f acc0 = {}, acc1 = {};
  for (int k0 = 0; k0 < D_MODEL; k0 += 32) {
    union { v16bf v; u32x4 q[2]; } A0, A1, Bm;
    // A (16x32 bf16): lanes 0-15 rows, K 0-7 / 16-23 ; lanes 16-31 K 8-15 / 24-31
    const __bf16* ap = &lds_x[l15 * XROW + k0 + lhalf * 8];
    A0.q[0] = *(const u32x4*)ap;
    A0.q[1] = *(const u32x4*)(ap + 16);
    A1.q[0] = *(const u32x4*)(ap + 16 * XROW);
    A1.q[1] = *(const u32x4*)(ap + 16 * XROW + 16);
    // B (32x16 bf16): lane = column; lanes 0-15 K 0-7 / 16-23 ; lanes 16-31 K 8-15 / 24-31
    const __bf16* bp = wrow + k0 + lhalf * 8;
    Bm.q[0] = *(const u32x4*)bp;
    Bm.q[1] = *(const u32x4*)(bp + 16);
    acc0 = __builtin_amdgcn_wmma_f32_16x16x32_bf16(
        false, A0.v, false, Bm.v, (short)0, acc0, false, false);
    acc1 = __builtin_amdgcn_wmma_f32_16x16x32_bf16(
        false, A1.v, false, Bm.v, (short)0, acc1, false, false);
  }

  // epilogue: C/D layout -> element (M = r + 8*lhalf [+16], N = lane&15)
  const int col = n0 + l15;
  #pragma unroll
  for (int half = 0; half < 2; ++half) {
    const v8f acc = half ? acc1 : acc0;
    #pragma unroll
    for (int r = 0; r < 8; ++r) {
      const size_t g = (size_t)(m0 + half * 16 + r + lhalf * 8);
      const float v = acc[r];
      if (mat == 0) {
        const float t = v + b_delta[col];                     // softplus
        ws_delta[g * NSTATE + col] = (t > 20.f) ? t : log1pf(__expf(t));
      } else if (mat == 1) {
        const float t = v + b_Bp[col];                        // u = Bt * x[:,:64]
        ws_u[g * NSTATE + col] = t * x[g * D_MODEL + col];
      } else {
        ws_ct[g * NSTATE + col] = v + b_Cp[col];
      }
    }
  }
}

// -------------------------------------------------------------------------
// Kernel 3: pass 1 of chunked scan. grid = B*NCHUNK = 256 workgroups.
//   thread owns h[i, j0..j0+16), i = tid/4, j0 = (tid%4)*16.
//   Emits chunk-local scan result hloc (from zero state) and chunk decay
//   P = exp(A_log * sum_t delta)  (product of per-step decays, 1 exp each).
// -------------------------------------------------------------------------
__global__ __launch_bounds__(256) void k_scan1(
    const float* __restrict__ ws_delta, const float* __restrict__ ws_u,
    const float* __restrict__ A_log,
    float* __restrict__ ws_P, float* __restrict__ ws_hloc) {
  __shared__ __align__(16) float d_l[CHUNK * NSTATE];
  __shared__ __align__(16) float u_l[CHUNK * NSTATE];
  const int tid = threadIdx.x;
  const int b = blockIdx.x >> 5, c = blockIdx.x & 31;
  const size_t g0 = ((size_t)b * SEQ + (size_t)c * CHUNK) * NSTATE;

  // async memory->LDS staging (ASYNCcnt path)
  {
    const unsigned od = lds_off(d_l), ou = lds_off(u_l);
    const float* gd = ws_delta + g0;
    const float* gu = ws_u + g0;
    for (int i4 = tid; i4 < CHUNK * NSTATE / 4; i4 += 256) {
      async_ld_b128(od + i4 * 16u, (unsigned long long)(gd + i4 * 4));
      async_ld_b128(ou + i4 * 16u, (unsigned long long)(gu + i4 * 4));
    }
    async_wait_all();
  }
  __syncthreads();

  const int row = tid >> 2, j0 = (tid & 3) * 16;
  float alog[16], h[16];
  #pragma unroll
  for (int j = 0; j < 16; ++j) {
    alog[j] = A_log[row * NSTATE + j0 + j];
    h[j] = 0.f;
  }
  float dsum = 0.f;
  for (int t = 0; t < CHUNK; ++t) {
    const float d = d_l[t * NSTATE + row];
    dsum += d;
    const float* up = &u_l[t * NSTATE + j0];
    #pragma unroll
    for (int j = 0; j < 16; ++j)
      h[j] = fmaf(__expf(alog[j] * d), h[j], up[j]);
  }
  float* Pp = ws_P    + (size_t)blockIdx.x * 4096 + tid * 16;
  float* Hp = ws_hloc + (size_t)blockIdx.x * 4096 + tid * 16;
  #pragma unroll
  for (int j = 0; j < 16; ++j) {
    Pp[j] = __expf(alog[j] * dsum);
    Hp[j] = h[j];
  }
}

// -------------------------------------------------------------------------
// Kernel 4: sequential cross-chunk combine (tiny): h' = P (.) h + hloc.
//   Stores the state entering each chunk. grid = 8, block = 256.
// -------------------------------------------------------------------------
__global__ __launch_bounds__(256) void k_combine(
    const float* __restrict__ ws_P, const float* __restrict__ ws_hloc,
    float* __restrict__ ws_hst) {
  const int b = blockIdx.x, tid = threadIdx.x;
  float h[16];
  #pragma unroll
  for (int j = 0; j < 16; ++j) h[j] = 0.f;
  for (int c = 0; c < NCHUNK; ++c) {
    const size_t base = ((size_t)b * NCHUNK + c) * 4096 + tid * 16;
    #pragma unroll
    for (int j = 0; j < 16; ++j) ws_hst[base + j] = h[j];
    #pragma unroll
    for (int j = 0; j < 16; ++j)
      h[j] = fmaf(ws_P[base + j], h[j], ws_hloc[base + j]);
  }
}

// -------------------------------------------------------------------------
// Kernel 5: pass 2 — replay each chunk from its true start state, emit y.
//   y[i] = sum_j C_t[j] * h[i,j]; 4-thread reduce via shfl_xor (wave32).
// -------------------------------------------------------------------------
__global__ __launch_bounds__(256) void k_scan2(
    const float* __restrict__ ws_delta, const float* __restrict__ ws_u,
    const float* __restrict__ ws_ct, const float* __restrict__ A_log,
    const float* __restrict__ ws_hst, float* __restrict__ out) {
  __shared__ __align__(16) float d_l[CHUNK * NSTATE];
  __shared__ __align__(16) float u_l[CHUNK * NSTATE];
  __shared__ __align__(16) float c_l[CHUNK * NSTATE];
  const int tid = threadIdx.x;
  const int b = blockIdx.x >> 5, c = blockIdx.x & 31;
  const size_t g0 = ((size_t)b * SEQ + (size_t)c * CHUNK) * NSTATE;

  // async memory->LDS staging (ASYNCcnt path)
  {
    const unsigned od = lds_off(d_l), ou = lds_off(u_l), oc = lds_off(c_l);
    const float* gd = ws_delta + g0;
    const float* gu = ws_u + g0;
    const float* gc = ws_ct + g0;
    for (int i4 = tid; i4 < CHUNK * NSTATE / 4; i4 += 256) {
      async_ld_b128(od + i4 * 16u, (unsigned long long)(gd + i4 * 4));
      async_ld_b128(ou + i4 * 16u, (unsigned long long)(gu + i4 * 4));
      async_ld_b128(oc + i4 * 16u, (unsigned long long)(gc + i4 * 4));
    }
    async_wait_all();
  }
  __syncthreads();

  const int row = tid >> 2, jb = tid & 3, j0 = jb * 16;
  float alog[16], h[16];
  const size_t hb = (size_t)blockIdx.x * 4096 + tid * 16;
  #pragma unroll
  for (int j = 0; j < 16; ++j) {
    alog[j] = A_log[row * NSTATE + j0 + j];
    h[j] = ws_hst[hb + j];
  }

  for (int t = 0; t < CHUNK; ++t) {
    const float d = d_l[t * NSTATE + row];
    const float* up = &u_l[t * NSTATE + j0];
    const float* cp = &c_l[t * NSTATE + j0];
    float y = 0.f;
    #pragma unroll
    for (int j = 0; j < 16; ++j) {
      h[j] = fmaf(__expf(alog[j] * d), h[j], up[j]);
      y = fmaf(cp[j], h[j], y);
    }
    y += __shfl_xor(y, 1, 32);
    y += __shfl_xor(y, 2, 32);
    if (jb == 0) out[g0 + (size_t)t * NSTATE + row] = y;  // [B,S,1,N] flat
  }
}

// -------------------------------------------------------------------------
extern "C" void kernel_launch(void* const* d_in, const int* in_sizes, int n_in,
                              void* d_out, int out_size, void* d_ws,
                              size_t ws_size, hipStream_t stream) {
  (void)in_sizes; (void)n_in; (void)out_size; (void)ws_size;
  const float* x       = (const float*)d_in[0];
  // d_in[1..6] = W/b for cog/beh/env: dead code in the reference, skipped.
  const float* W_delta = (const float*)d_in[7];
  const float* b_delta = (const float*)d_in[8];
  const float* W_Bp    = (const float*)d_in[9];
  const float* b_Bp    = (const float*)d_in[10];
  const float* W_Cp    = (const float*)d_in[11];
  const float* b_Cp    = (const float*)d_in[12];
  const float* A_log   = (const float*)d_in[13];

  float*  ws   = (float*)d_ws;
  float*  wsD  = ws + WS_DELTA;
  float*  wsU  = ws + WS_U;
  float*  wsC  = ws + WS_CT;
  float*  wsP  = ws + WS_P;
  float*  wsH  = ws + WS_HLOC;
  float*  wsS  = ws + WS_HST;
  __bf16* wbf  = (__bf16*)(ws + WS_WBF);

  k_cvt_w  <<<768,  256, 0, stream>>>(W_delta, W_Bp, W_Cp, wbf);
  k_gemm   <<<M_TOTAL / M_TILE, 384, M_TILE * XROW * sizeof(__bf16), stream>>>(
      x, wbf, b_delta, b_Bp, b_Cp, wsD, wsU, wsC);
  k_scan1  <<<BATCH * NCHUNK, 256, 0, stream>>>(wsD, wsU, A_log, wsP, wsH);
  k_combine<<<BATCH, 256, 0, stream>>>(wsP, wsH, wsS);
  k_scan2  <<<BATCH * NCHUNK, 256, 0, stream>>>(wsD, wsU, wsC, A_log, wsS,
                                                (float*)d_out);
}
